// metric_learning_8_25443386262193
// MI455X (gfx1250) — compile-verified, matrix operand order checked by reading, not run on hardware
//
#include <hip/hip_runtime.h>

// ---------------- constants ----------------
#define NROW 8192      // B*L
#define DM   128       // embedding dim
#define HD   256       // hidden dim (2*D)
#define LOG2L 9        // L = 512
#define BN_EPS 1e-5f

typedef __bf16 bf16_t;
typedef __attribute__((ext_vector_type(16))) __bf16 v16bf;
typedef __attribute__((ext_vector_type(8)))  __bf16 v8bf;
typedef __attribute__((ext_vector_type(8)))  float  v8f;

union FragU { v16bf v; v8bf h[2]; };

// Load a 16x32 bf16 WMMA A-fragment (or 32x16 B-fragment, same per-lane packing)
// from a row-major matrix. base_row must already point at the start of the row
// (m0 + (lane&15)) for A, or (n0 + (lane&15)) for B. kb = K-chunk base (mult of 32).
// Packing: half element e -> K = 2*(e&8) + 8*(lane>>4) + (e&7).
__device__ __forceinline__ v16bf load_frag(const bf16_t* base_row, int kb, int lane) {
    int hi = (lane >> 4) & 1;
    FragU f;
    f.h[0] = *(const v8bf*)(base_row + kb + 8 * hi);        // e=0..7 : K = kb + 8*hi + e
    f.h[1] = *(const v8bf*)(base_row + kb + 16 + 8 * hi);   // e=8..15: K = kb + 16 + 8*hi + (e-8)
    return f.v;
}

// ---------------- kernel 0: init ----------------
__global__ void init_kernel(float* denom, float* out) {
    int i = blockIdx.x * blockDim.x + threadIdx.x;
    if (i < NROW) denom[i] = 0.0f;
    if (i == 0) out[0] = 0.0f;
}

// ---------------- kernel: f32 -> bf16 convert ----------------
__global__ void cvt_kernel(const float* src, bf16_t* dst, int n) {
    int i = blockIdx.x * blockDim.x + threadIdx.x;
    if (i < n) dst[i] = (bf16_t)src[i];
}

// ---------------- kernel: GEMM1  h = relu(x @ w1^T + b1) ----------------
// x: [2][N][128] bf16, w1: [256][128] bf16, h: [2][N][256] bf16
// grid (N/16, 1, 2), block 256 (8 waves). Wave w does col tiles w and w+8.
__global__ void gemm1_kernel(const bf16_t* __restrict__ xb, const bf16_t* __restrict__ w1b,
                             const float* __restrict__ b1, bf16_t* __restrict__ h) {
    int z    = blockIdx.z;
    int tid  = threadIdx.x;
    int wave = tid >> 5, lane = tid & 31, hi = lane >> 4;
    int m0   = blockIdx.x * 16;
    const bf16_t* x    = xb + (size_t)z * NROW * DM;
    bf16_t*       hout = h  + (size_t)z * NROW * HD;

    const bf16_t* arow = x + (size_t)(m0 + (lane & 15)) * DM;
    v16bf a[4];
#pragma unroll
    for (int kc = 0; kc < 4; ++kc) a[kc] = load_frag(arow, 32 * kc, lane);

#pragma unroll
    for (int t = 0; t < 2; ++t) {
        int n0 = 16 * (wave + 8 * t);
        const bf16_t* brow = w1b + (size_t)(n0 + (lane & 15)) * DM;
        v8f c = {};
#pragma unroll
        for (int kc = 0; kc < 4; ++kc) {
            v16bf b = load_frag(brow, 32 * kc, lane);
            c = __builtin_amdgcn_wmma_f32_16x16x32_bf16(false, a[kc], false, b,
                                                        (short)0, c, false, false);
        }
        int   col  = n0 + (lane & 15);
        float bias = b1[col];
#pragma unroll
        for (int r = 0; r < 8; ++r) {
            float v = c[r] + bias;
            v = v > 0.0f ? v : 0.0f;   // ReLU
            hout[(size_t)(m0 + r + 8 * hi) * HD + col] = (bf16_t)v;
        }
    }
}

// ---------------- kernel: GEMM2  y = h @ w2^T + b2 ----------------
// h: [2][N][256] bf16, w2: [128][256] bf16, y: [2][N][128] f32
// grid (N/16, 1, 2), block 256 (8 waves -> 8 col tiles of 16 = 128 cols).
__global__ void gemm2_kernel(const bf16_t* __restrict__ h, const bf16_t* __restrict__ w2b,
                             const float* __restrict__ b2, float* __restrict__ y) {
    int z    = blockIdx.z;
    int tid  = threadIdx.x;
    int wave = tid >> 5, lane = tid & 31, hi = lane >> 4;
    int m0   = blockIdx.x * 16;
    const bf16_t* hin  = h + (size_t)z * NROW * HD;
    float*        yout = y + (size_t)z * NROW * DM;

    const bf16_t* arow = hin + (size_t)(m0 + (lane & 15)) * HD;
    int n0 = 16 * wave;
    const bf16_t* brow = w2b + (size_t)(n0 + (lane & 15)) * HD;

    v8f c = {};
#pragma unroll
    for (int kc = 0; kc < 8; ++kc) {
        v16bf a = load_frag(arow, 32 * kc, lane);
        v16bf b = load_frag(brow, 32 * kc, lane);
        c = __builtin_amdgcn_wmma_f32_16x16x32_bf16(false, a, false, b,
                                                    (short)0, c, false, false);
    }
    int   col  = n0 + (lane & 15);
    float bias = b2[col];
#pragma unroll
    for (int r = 0; r < 8; ++r)
        yout[(size_t)(m0 + r + 8 * hi) * DM + col] = c[r] + bias;
}

// ---------------- kernel: BN column stats ----------------
// grid (128, 2), block 256. mu/rsig: [2][128]
__global__ void bn_stats_kernel(const float* __restrict__ y, float* __restrict__ mu,
                                float* __restrict__ rsig) {
    int c = blockIdx.x, z = blockIdx.y, t = threadIdx.x;
    const float* yb = y + (size_t)z * NROW * DM;
    float s = 0.0f, s2 = 0.0f;
    for (int r = t; r < NROW; r += 256) {
        float v = yb[(size_t)r * DM + c];
        s += v; s2 += v * v;
    }
    __shared__ float ls[256], ls2[256];
    ls[t] = s; ls2[t] = s2; __syncthreads();
    for (int stride = 128; stride > 0; stride >>= 1) {
        if (t < stride) { ls[t] += ls[t + stride]; ls2[t] += ls2[t + stride]; }
        __syncthreads();
    }
    if (t == 0) {
        float m   = ls[0] / (float)NROW;
        float var = ls2[0] / (float)NROW - m * m;
        mu[z * DM + c]   = m;
        rsig[z * DM + c] = rsqrtf(var + BN_EPS);
    }
}

// ---------------- kernel: BN apply + L2 normalize -> bf16 ----------------
// grid (N, 2), block 128.
__global__ void bn_norm_kernel(const float* __restrict__ y, const float* __restrict__ mu,
                               const float* __restrict__ rsig, const float* __restrict__ gamma,
                               const float* __restrict__ beta, bf16_t* __restrict__ zb) {
    int i = blockIdx.x, z = blockIdx.y, t = threadIdx.x;
    const float* yrow = y + (size_t)z * NROW * DM + (size_t)i * DM;
    float v = (yrow[t] - mu[z * DM + t]) * rsig[z * DM + t] * gamma[t] + beta[t];
    __shared__ float red[128];
    red[t] = v * v; __syncthreads();
    for (int s = 64; s > 0; s >>= 1) {
        if (t < s) red[t] += red[t + s];
        __syncthreads();
    }
    float inv = rsqrtf(red[0]);
    zb[(size_t)z * NROW * DM + (size_t)i * DM + t] = (bf16_t)(v * inv);
}

// ---------------- kernel: per-row prep (diag, s_next, two, pos_sum) ----------------
__global__ void prep_kernel(const bf16_t* __restrict__ z1b, const bf16_t* __restrict__ z2b,
                            const int* __restrict__ ids, int* __restrict__ idsn,
                            int* __restrict__ twoArr, float* __restrict__ possum) {
    int i = blockIdx.x * blockDim.x + threadIdx.x;
    if (i >= NROW) return;
    int in1 = (i + 1) & (NROW - 1);
    idsn[i] = ids[in1];
    int two = (i < NROW - 1) && ((i >> LOG2L) == ((i + 1) >> LOG2L));
    twoArr[i] = two;
    const bf16_t* a  = z1b + (size_t)i * DM;
    const bf16_t* b  = z2b + (size_t)i * DM;
    const bf16_t* bn = z2b + (size_t)in1 * DM;
    float d = 0.0f, sn = 0.0f;
#pragma unroll 4
    for (int k = 0; k < DM; ++k) {
        float av = (float)a[k];
        d  += av * (float)b[k];
        sn += av * (float)bn[k];
    }
    possum[i] = d + (two ? sn : 0.0f);
}

// ---------------- kernel: fused S-tile WMMA + mask + exp + denominator ----------------
// grid (N/16/8 = 64, JCHUNKS), block 256 (8 waves, each wave owns a 16-row strip).
#define JCHUNKS 8
__global__ void loss_denom_kernel(const bf16_t* __restrict__ z1b, const bf16_t* __restrict__ z2b,
                                  const int* __restrict__ ids, const int* __restrict__ idsn,
                                  const int* __restrict__ twoArr, float* __restrict__ denom) {
    int tid  = threadIdx.x;
    int wave = tid >> 5, lane = tid & 31, hi = lane >> 4;
    int m0   = (blockIdx.x * 8 + wave) * 16;
    const int JC = NROW / JCHUNKS;
    int jbeg = blockIdx.y * JC, jend = jbeg + JC;

    // Preload A fragments for this wave's 16 rows (K = 0..127 in 4 chunks).
    const bf16_t* arow = z1b + (size_t)(m0 + (lane & 15)) * DM;
    v16bf a[4];
#pragma unroll
    for (int kc = 0; kc < 4; ++kc) a[kc] = load_frag(arow, 32 * kc, lane);

    // Row metadata for this lane's 8 rows (row = m0 + 8*hi + r).
    int ri[8], rn[8], rs[8], rt[8];
#pragma unroll
    for (int r = 0; r < 8; ++r) {
        int i = m0 + 8 * hi + r;
        ri[r] = ids[i];
        rn[r] = idsn[i];
        rs[r] = i >> LOG2L;
        rt[r] = twoArr[i];
    }

    float acc[8];
#pragma unroll
    for (int r = 0; r < 8; ++r) acc[r] = 0.0f;

    for (int j0 = jbeg; j0 < jend; j0 += 16) {
        const bf16_t* brow = z2b + (size_t)(j0 + (lane & 15)) * DM;
        v8f c = {};
#pragma unroll
        for (int kc = 0; kc < 4; ++kc) {
            v16bf b = load_frag(brow, 32 * kc, lane);
            c = __builtin_amdgcn_wmma_f32_16x16x32_bf16(false, a[kc], false, b,
                                                        (short)0, c, false, false);
        }
        int j   = j0 + (lane & 15);
        int idj = ids[j];
        int sj  = j >> LOG2L;
#pragma unroll
        for (int r = 0; r < 8; ++r) {
            int  i    = m0 + 8 * hi + r;
            bool neg1 = (idj != ri[r]);
            bool m    = rt[r] ? (neg1 && (idj != rn[r]) && (sj != rs[r])) : neg1;
            m = m || (j == i) || (rt[r] && (j == i + 1));
            if (m) acc[r] += __expf(c[r]);   // TEMP = 1
        }
    }

    // Reduce each row's partial across the 16 column-lanes of its half.
#pragma unroll
    for (int r = 0; r < 8; ++r) {
        acc[r] += __shfl_xor(acc[r], 1, 32);
        acc[r] += __shfl_xor(acc[r], 2, 32);
        acc[r] += __shfl_xor(acc[r], 4, 32);
        acc[r] += __shfl_xor(acc[r], 8, 32);
    }
#pragma unroll
    for (int r = 0; r < 8; ++r)
        if ((lane & 15) == r)
            atomicAdd(&denom[m0 + 8 * hi + r], acc[r]);
}

// ---------------- kernel: final loss reduction ----------------
__global__ void final_kernel(const float* __restrict__ possum, const float* __restrict__ denom,
                             const int* __restrict__ twoArr, float* __restrict__ out) {
    int i = blockIdx.x * blockDim.x + threadIdx.x;
    int t = threadIdx.x;
    float np    = 1.0f + (float)twoArr[i];
    float inner = -(possum[i] - __logf(denom[i])) / np;
    __shared__ float red[256];
    red[t] = inner; __syncthreads();
    for (int s = 128; s > 0; s >>= 1) {
        if (t < s) red[t] += red[t + s];
        __syncthreads();
    }
    if (t == 0) atomicAdd(out, red[0] / (float)NROW);
}

// ---------------- host launcher ----------------
extern "C" void kernel_launch(void* const* d_in, const int* in_sizes, int n_in,
                              void* d_out, int out_size, void* d_ws, size_t ws_size,
                              hipStream_t stream) {
    const int*   ids   = (const int*)d_in[0];
    // d_in[1] = log_mask (unused by reference math)
    const float* x1    = (const float*)d_in[2];
    const float* x2    = (const float*)d_in[3];
    const float* w1    = (const float*)d_in[4];
    const float* b1    = (const float*)d_in[5];
    const float* w2    = (const float*)d_in[6];
    const float* b2    = (const float*)d_in[7];
    const float* gamma = (const float*)d_in[8];
    const float* beta  = (const float*)d_in[9];
    float* out = (float*)d_out;

    // workspace carve-up (256B aligned)
    char* ws = (char*)d_ws;
    size_t o = 0;
    auto carve = [&](size_t bytes) { char* p = ws + o; o = (o + bytes + 255) & ~(size_t)255; return p; };
    bf16_t* xb   = (bf16_t*)carve((size_t)2 * NROW * DM * sizeof(bf16_t)); // [2][N][128]
    bf16_t* w1b  = (bf16_t*)carve((size_t)HD * DM * sizeof(bf16_t));       // [256][128]
    bf16_t* w2b  = (bf16_t*)carve((size_t)DM * HD * sizeof(bf16_t));       // [128][256]
    bf16_t* hbuf = (bf16_t*)carve((size_t)2 * NROW * HD * sizeof(bf16_t)); // [2][N][256]
    float*  ybuf = (float*) carve((size_t)2 * NROW * DM * sizeof(float));  // [2][N][128]
    float*  mu   = (float*) carve((size_t)2 * DM * sizeof(float));
    float*  rsig = (float*) carve((size_t)2 * DM * sizeof(float));
    bf16_t* zb   = (bf16_t*)carve((size_t)2 * NROW * DM * sizeof(bf16_t)); // [2][N][128]
    float*  denom  = (float*)carve((size_t)NROW * sizeof(float));
    float*  possum = (float*)carve((size_t)NROW * sizeof(float));
    int*    idsn   = (int*)  carve((size_t)NROW * sizeof(int));
    int*    twoArr = (int*)  carve((size_t)NROW * sizeof(int));
    (void)ws_size; (void)in_sizes; (void)n_in; (void)out_size;

    bf16_t* z1b = zb;
    bf16_t* z2b = zb + (size_t)NROW * DM;

    // 0) init accumulators
    init_kernel<<<dim3(NROW / 256), dim3(256), 0, stream>>>(denom, out);

    // 1) f32 -> bf16 conversions
    cvt_kernel<<<dim3((NROW * DM + 255) / 256), dim3(256), 0, stream>>>(x1, xb, NROW * DM);
    cvt_kernel<<<dim3((NROW * DM + 255) / 256), dim3(256), 0, stream>>>(x2, xb + (size_t)NROW * DM, NROW * DM);
    cvt_kernel<<<dim3((HD * DM + 255) / 256), dim3(256), 0, stream>>>(w1, w1b, HD * DM);
    cvt_kernel<<<dim3((DM * HD + 255) / 256), dim3(256), 0, stream>>>(w2, w2b, DM * HD);

    // 2) MLP
    gemm1_kernel<<<dim3(NROW / 16, 1, 2), dim3(256), 0, stream>>>(xb, w1b, b1, hbuf);
    gemm2_kernel<<<dim3(NROW / 16, 1, 2), dim3(256), 0, stream>>>(hbuf, w2b, b2, ybuf);

    // 3) BatchNorm (train-mode batch stats) + L2 normalize -> bf16 z
    bn_stats_kernel<<<dim3(DM, 2), dim3(256), 0, stream>>>(ybuf, mu, rsig);
    bn_norm_kernel<<<dim3(NROW, 2), dim3(128), 0, stream>>>(ybuf, mu, rsig, gamma, beta, zb);

    // 4) per-row prep (diag, s_next, two flag, pos_sum)
    prep_kernel<<<dim3(NROW / 256), dim3(256), 0, stream>>>(z1b, z2b, ids, idsn, twoArr, possum);

    // 5) fused S = z1@z2^T tile + mask + exp + denominator accumulation
    loss_denom_kernel<<<dim3(NROW / 16 / 8, JCHUNKS), dim3(256), 0, stream>>>(
        z1b, z2b, ids, idsn, twoArr, denom);

    // 6) final scalar loss
    final_kernel<<<dim3(NROW / 256), dim3(256), 0, stream>>>(possum, denom, twoArr, out);
}
